// EquiSchNet_64939905515670
// MI455X (gfx1250) — matrix operand order repaired and checked by reading, not compiled, not run on hardware
//
#include <hip/hip_runtime.h>
#include <hip/hip_bf16.h>

// ---------------------------------------------------------------------------
// EquiSchNet forward on gfx1250 (CDNA5), wave32 + WMMA bf16.
//
// Model constants (match reference):
#define M1   2000
#define M2   400
#define BS   8
#define AT   2400            // M1+M2
#define NTOT (BS*AT)         // 19200 nodes
#define Hh   64
#define Ff   64
#define NG   50
#define Ll   4
#define Cc   4
#define E1   60000
#define E2   20000
#define E3   20000
#define LOG2F_SSP 0.6931471805599453f

typedef __attribute__((ext_vector_type(16))) __bf16 v16bf;
typedef __attribute__((ext_vector_type(8)))  float  v8f;

// ---------------------------------------------------------------------------
// helpers

__device__ __forceinline__ float ssp(float x) {
    // shifted softplus: log(1+exp(x)) - log(2), overflow-safe
    float sp = (x > 20.0f) ? x : log1pf(__expf(x));
    return sp - LOG2F_SSP;
}

// A-matrix fragment: 16x32 bf16 slice (rows m=0..15, K = kk..kk+31) from a
// 16x64 row-major bf16 LDS tile.  ISA 7.12.2 layout:
//   lane<16 : M=lane,   K-groups {0..7, 16..23} (+kk)
//   lane>=16: M=lane-16,K-groups {8..15,24..31} (+kk)
__device__ __forceinline__ v16bf load_A_frag(const __bf16* A, int kk, int lane) {
    const int m  = lane & 15;
    const int ko = (lane >> 4) << 3;   // 0 or 8
    v16bf a;
#pragma unroll
    for (int j = 0; j < 16; ++j) {
        const int k = kk + ko + j + ((j >> 3) << 3);   // j<8 -> j ; j>=8 -> j+8
        a[j] = A[m * 64 + k];
    }
    return a;
}

// B-matrix fragment: 32x16 bf16 slice (N = nn + (lane&15), K-groups as A) from a
// 64x64 row-major bf16 LDS weight tile: lane holds one column of B.
__device__ __forceinline__ v16bf load_B_frag(const __bf16* B, int kk, int nn, int lane) {
    const int n  = nn + (lane & 15);
    const int ko = (lane >> 4) << 3;   // 0 or 8
    v16bf b;
#pragma unroll
    for (int j = 0; j < 16; ++j) b[j] = B[(kk + ko + j + ((j >> 3) << 3)) * 64 + n];
    return b;
}

// acc[nt] element r  <->  C[m = r + 8*(lane>=16)][n = nt*16 + (lane&15)]
// 16x64 = 16x64 @ 64x64, all in bf16 LDS, f32 accumulate.
__device__ __forceinline__ void gemm_16x64(const __bf16* A, const __bf16* B,
                                           int lane, v8f acc[4]) {
#pragma unroll
    for (int kk = 0; kk < 64; kk += 32) {
        v16bf a = load_A_frag(A, kk, lane);
#pragma unroll
        for (int nt = 0; nt < 4; ++nt) {
            v16bf b = load_B_frag(B, kk, nt * 16, lane);
            acc[nt] = __builtin_amdgcn_wmma_f32_16x16x32_bf16(
                false, a, false, b, (short)0, acc[nt], false, false);
        }
    }
}

// ---------------------------------------------------------------------------
// 0) initial embedding: h[b*AT+i][f]
__global__ void embed_kernel(const int* __restrict__ sites,
                             const int* __restrict__ sites_p,
                             const float* __restrict__ emb_w,
                             const float* __restrict__ emb_p_w,
                             float* __restrict__ h) {
    int t = blockIdx.x * blockDim.x + threadIdx.x;
    if (t >= NTOT * Hh) return;
    int f = t & 63, n = t >> 6;
    int b = n / AT, i = n - b * AT;
    float v;
    if (i < M1) v = emb_w[sites[b * M1 + i] * Hh + f];
    else        v = emb_p_w[sites_p[b * M2 + (i - M1)] * Hh + f];
    h[t] = v;
}

// ---------------------------------------------------------------------------
// 1) edge-filter: W[e,:] = colored_mlp(smear(ew[e]))[col[e]] * cos_cutoff(ew[e])
//    Computed once per (layer,graph) -- identical across batch replicas.
//    One wave per 16-edge tile; 8 tiles per 256-thread block.
__global__ __launch_bounds__(256)
void filter_kernel(const float* __restrict__ ew, const int* __restrict__ col,
                   const float* __restrict__ w1, const float* __restrict__ b1,
                   const float* __restrict__ w2, const float* __restrict__ b2,
                   float* __restrict__ Wout, int E) {
    __shared__ __bf16 sW1[64 * 64];          // w1[c] K-padded 50->64  (8 KB)
    __shared__ __bf16 sW2[64 * 64];          // w2[c]                  (8 KB)
    __shared__ __bf16 sA1[8][16 * 64];       // smeared attrs          (16 KB)
    __shared__ __bf16 sA2[8][16 * 64];       // hidden activations     (16 KB)
    __shared__ float  sB1[64], sB2[64];

    const int tid  = threadIdx.x;
    const int wave = tid >> 5, lane = tid & 31;
    const int e0   = (blockIdx.x * 8 + wave) * 16;   // wave's first edge

    // Gaussian smearing into sA1 (zero-pad g>=50 and e>=E)
    const float step  = 10.0f / 49.0f;
    const float coeff = -0.5f / (step * step);
    for (int i = lane; i < 16 * 64; i += 32) {
        int m = i >> 6, g = i & 63;
        int e = e0 + m;
        float v = 0.0f;
        if (e < E && g < NG) {
            float d = ew[e] - (float)g * step;
            v = __expf(coeff * d * d);
        }
        sA1[wave][i] = (__bf16)v;
    }

    // per-row (edge) color and cosine cutoff, registers
    int   cm[8];
    float ccut[8];
#pragma unroll
    for (int r = 0; r < 8; ++r) {
        int e = e0 + r + ((lane >> 4) << 3);
        cm[r]   = (e < E) ? col[e] : -1;
        float d = (e < E) ? ew[e] : 0.0f;
        ccut[r] = 0.5f * (__cosf(d * 0.3141592653589793f) + 1.0f);  // pi/CUTOFF
    }

    float outv[4][8];
#pragma unroll
    for (int nt = 0; nt < 4; ++nt)
#pragma unroll
        for (int r = 0; r < 8; ++r) outv[nt][r] = 0.0f;

    for (int c = 0; c < Cc; ++c) {
        __syncthreads();
        for (int i = tid; i < 64 * 64; i += 256) {
            int k = i >> 6, n = i & 63;
            sW1[i] = (__bf16)((k < NG) ? w1[(c * NG + k) * 64 + n] : 0.0f);
            sW2[i] = (__bf16)w2[(c * 64 + k) * 64 + n];
        }
        if (tid < 64) { sB1[tid] = b1[c * 64 + tid]; sB2[tid] = b2[c * 64 + tid]; }
        __syncthreads();

        // layer 1: attr @ w1[c] + b1 -> ssp -> sA2
        v8f acc[4] = {};
        gemm_16x64(sA1[wave], sW1, lane, acc);
#pragma unroll
        for (int nt = 0; nt < 4; ++nt) {
            float bv = sB1[nt * 16 + (lane & 15)];
#pragma unroll
            for (int r = 0; r < 8; ++r) {
                int m = r + ((lane >> 4) << 3);
                int n = nt * 16 + (lane & 15);
                sA2[wave][m * 64 + n] = (__bf16)ssp(acc[nt][r] + bv);
            }
        }
        // same-wave LDS RAW (DS ops are in-order within a wave)

        // layer 2: h1 @ w2[c] + b2, select rows of this color
        v8f acc2[4] = {};
        gemm_16x64(sA2[wave], sW2, lane, acc2);
#pragma unroll
        for (int nt = 0; nt < 4; ++nt) {
            float bv = sB2[nt * 16 + (lane & 15)];
#pragma unroll
            for (int r = 0; r < 8; ++r)
                if (cm[r] == c) outv[nt][r] = acc2[nt][r] + bv;
        }
    }

    // apply cutoff, store
#pragma unroll
    for (int nt = 0; nt < 4; ++nt)
#pragma unroll
        for (int r = 0; r < 8; ++r) {
            int e = e0 + r + ((lane >> 4) << 3);
            if (e < E)
                Wout[e * 64 + nt * 16 + (lane & 15)] = outv[nt][r] * ccut[r];
        }
}

// ---------------------------------------------------------------------------
// 2) generic node GEMM: out = [ssp?](in) @ w64x64 [+ bias] [accumulated]
//    flags: 1 = add bias, 2 = ssp on input, 4 = accumulate into out
__global__ __launch_bounds__(256)
void gemm64_kernel(const float* __restrict__ in, const float* __restrict__ w,
                   const float* __restrict__ bias, float* __restrict__ out,
                   int n_rows, int flags) {
    __shared__ __bf16 sW[64 * 64];
    __shared__ __bf16 sA[8][16 * 64];
    __shared__ float  sB[64];
    const int tid = threadIdx.x, wave = tid >> 5, lane = tid & 31;

    for (int i = tid; i < 64 * 64; i += 256) sW[i] = (__bf16)w[i];
    if (tid < 64) sB[tid] = (flags & 1) ? bias[tid] : 0.0f;

    const int r0 = (blockIdx.x * 8 + wave) * 16;
    for (int i = lane; i < 16 * 64; i += 32) {
        int m = i >> 6, n = i & 63;
        int row = r0 + m;
        float v = (row < n_rows) ? in[row * 64 + n] : 0.0f;
        if (flags & 2) v = ssp(v);
        sA[wave][i] = (__bf16)v;
    }
    __syncthreads();

    v8f acc[4] = {};
    gemm_16x64(sA[wave], sW, lane, acc);

#pragma unroll
    for (int nt = 0; nt < 4; ++nt) {
        const int n  = nt * 16 + (lane & 15);
        const float bv = sB[n];
#pragma unroll
        for (int r = 0; r < 8; ++r) {
            int row = r0 + r + ((lane >> 4) << 3);
            if (row < n_rows) {
                float v = acc[nt][r] + bv;
                if (flags & 4) out[row * 64 + n] += v;
                else           out[row * 64 + n]  = v;
            }
        }
    }
}

// ---------------------------------------------------------------------------
// 3) gather-modulate-scatter: agg[dst+b*AT] += x[src+b*AT] * W[e]
//    one thread per (b, e, 4-feature group)
__global__ void scatter_kernel(const float* __restrict__ x,
                               const float* __restrict__ Wf,
                               const int* __restrict__ ei,
                               float* __restrict__ agg, int E) {
    long long t = (long long)blockIdx.x * blockDim.x + threadIdx.x;
    if (t >= (long long)BS * E * 16) return;
    int q = (int)(t & 15);
    long long be = t >> 4;
    int e = (int)(be % E);
    int b = (int)(be / E);
    int src = ei[2 * e]     + b * AT;
    int dst = ei[2 * e + 1] + b * AT;
    const float4 wv = *(const float4*)(Wf + (size_t)e * 64 + q * 4);
    const float4 xv = *(const float4*)(x  + (size_t)src * 64 + q * 4);
    float* d = agg + (size_t)dst * 64 + q * 4;
    atomicAdd(d + 0, xv.x * wv.x);
    atomicAdd(d + 1, xv.y * wv.y);
    atomicAdd(d + 2, xv.z * wv.z);
    atomicAdd(d + 3, xv.w * wv.w);
}

__global__ void zero_kernel(float* __restrict__ p, int n) {
    int t = blockIdx.x * blockDim.x + threadIdx.x;
    if (t < n) p[t] = 0.0f;
}

__global__ void add_kernel(float* __restrict__ a, const float* __restrict__ b, int n) {
    int t = blockIdx.x * blockDim.x + threadIdx.x;
    if (t < n) a[t] += b[t];
}

// ---------------------------------------------------------------------------
// 4) readout: out[b] = sum_pore ((h@W1+b1)@W2+b2)
//           = (sum_pore h) . v + M2*(b1.w2 + b2),  v[f] = sum_j W1[f,j]*w2[j]
__global__ __launch_bounds__(256)
void readout_kernel(const float* __restrict__ h,
                    const float* __restrict__ w1, const float* __restrict__ b1,
                    const float* __restrict__ w2, const float* __restrict__ b2,
                    float* __restrict__ out) {
    __shared__ float sv[64];
    __shared__ float red[256];
    const int b = blockIdx.x, tid = threadIdx.x;
    if (tid < 64) {
        float a = 0.0f;
        for (int j = 0; j < 32; ++j) a += w1[tid * 32 + j] * w2[j];
        sv[tid] = a;
    }
    __syncthreads();
    const int f = tid & 63, g = tid >> 6;          // 4 node-groups x 64 feats
    float acc = 0.0f;
    for (int i = g; i < M2; i += 4)
        acc += h[(size_t)(b * AT + M1 + i) * 64 + f];
    red[tid] = acc;
    __syncthreads();
    if (tid < 64)
        red[tid] = (red[tid] + red[tid + 64] + red[tid + 128] + red[tid + 192]) * sv[tid];
    __syncthreads();
    if (tid == 0) {
        float tot = 0.0f;
        for (int i = 0; i < 64; ++i) tot += red[i];
        float s = b2[0];
        for (int j = 0; j < 32; ++j) s += b1[j] * w2[j];
        out[b] = tot + (float)M2 * s;
    }
}

// ---------------------------------------------------------------------------

extern "C" void kernel_launch(void* const* d_in, const int* in_sizes, int n_in,
                              void* d_out, int out_size, void* d_ws, size_t ws_size,
                              hipStream_t stream) {
    (void)in_sizes; (void)n_in; (void)out_size; (void)ws_size;

    const int*   sites   = (const int*)d_in[0];
    const int*   sites_p = (const int*)d_in[1];
    const int*   ei[3]   = {(const int*)d_in[2],  (const int*)d_in[5],  (const int*)d_in[8]};
    const float* ew[3]   = {(const float*)d_in[3],(const float*)d_in[6],(const float*)d_in[9]};
    const int*   col[3]  = {(const int*)d_in[4],  (const int*)d_in[7],  (const int*)d_in[10]};
    const float* emb_w   = (const float*)d_in[11];
    const float* emb_p_w = (const float*)d_in[12];
    const float* mlp_w1  = (const float*)d_in[13];  // [L,3,C,50,64]
    const float* mlp_b1  = (const float*)d_in[14];  // [L,3,C,64]
    const float* mlp_w2  = (const float*)d_in[15];  // [L,3,C,64,64]
    const float* mlp_b2  = (const float*)d_in[16];  // [L,3,C,64]
    const float* conv_w1 = (const float*)d_in[17];  // [L,3,64,64]
    const float* conv_w2 = (const float*)d_in[18];  // [L,3,64,64]
    const float* conv_b2 = (const float*)d_in[19];  // [L,3,64]
    const float* blk_w   = (const float*)d_in[20];  // [L,3,64,64]
    const float* blk_b   = (const float*)d_in[21];  // [L,3,64]
    const float* out_w1  = (const float*)d_in[22];
    const float* out_b1  = (const float*)d_in[23];
    const float* out_w2  = (const float*)d_in[24];
    const float* out_b2  = (const float*)d_in[25];

    const int Eg[3] = {E1, E2, E3};

    // workspace layout (floats): h | x | agg | delta | Wfilter  (~35 MB total)
    float* ws    = (float*)d_ws;
    float* h     = ws;
    float* x     = h     + (size_t)NTOT * 64;
    float* agg   = x     + (size_t)NTOT * 64;
    float* delta = agg   + (size_t)NTOT * 64;
    float* Wf    = delta + (size_t)NTOT * 64;

    const int nElems    = NTOT * 64;
    const int gridElems = (nElems + 255) / 256;
    const int gridRows  = (NTOT + 127) / 128;   // 8 waves x 16 rows per block

    embed_kernel<<<gridElems, 256, 0, stream>>>(sites, sites_p, emb_w, emb_p_w, h);

    for (int l = 0; l < Ll; ++l) {
        for (int g = 0; g < 3; ++g) {
            const int E  = Eg[g];
            const int lg = l * 3 + g;

            // edge filter W[E,64] (shared by all batch replicas)
            filter_kernel<<<(E + 127) / 128, 256, 0, stream>>>(
                ew[g], col[g],
                mlp_w1 + (size_t)lg * Cc * NG * 64, mlp_b1 + (size_t)lg * Cc * 64,
                mlp_w2 + (size_t)lg * Cc * 64 * 64, mlp_b2 + (size_t)lg * Cc * 64,
                Wf, E);

            // x = h @ cw1
            gemm64_kernel<<<gridRows, 256, 0, stream>>>(
                h, conv_w1 + (size_t)lg * 64 * 64, nullptr, x, NTOT, 0);

            // agg = 0 ; agg[dst] += x[src] * W[e]
            zero_kernel<<<gridElems, 256, 0, stream>>>(agg, nElems);
            long long work = (long long)BS * E * 16;
            scatter_kernel<<<(unsigned)((work + 255) / 256), 256, 0, stream>>>(
                x, Wf, ei[g], agg, E);

            // x <- agg @ cw2 + cb2
            gemm64_kernel<<<gridRows, 256, 0, stream>>>(
                agg, conv_w2 + (size_t)lg * 64 * 64, conv_b2 + (size_t)lg * 64,
                x, NTOT, 1);

            // delta (+)= ssp(x) @ bw + bb
            gemm64_kernel<<<gridRows, 256, 0, stream>>>(
                x, blk_w + (size_t)lg * 64 * 64, blk_b + (size_t)lg * 64,
                delta, NTOT, 1 | 2 | (g ? 4 : 0));
        }
        add_kernel<<<gridElems, 256, 0, stream>>>(h, delta, nElems);
    }

    readout_kernel<<<BS, 256, 0, stream>>>(h, out_w1, out_b1, out_w2, out_b2,
                                           (float*)d_out);
}
